// VFNet_17403207483938
// MI455X (gfx1250) — matrix operand order repaired and checked by reading, not compiled, not run on hardware
//
#include <hip/hip_runtime.h>

typedef unsigned short u16;
typedef unsigned int   u32;
typedef unsigned long long u64;

typedef __attribute__((ext_vector_type(16))) __bf16 v16bf;
typedef __attribute__((ext_vector_type(8)))  float  v8f;
typedef __attribute__((ext_vector_type(4)))  u32    v4u;

union Frag16 { v16bf bf; v4u q[2]; };

#ifndef USE_ASYNC
#define USE_ASYNC 1
#endif

__device__ __forceinline__ u16 f2bf(float f) {
    u32 u = __float_as_uint(f);
    u32 r = u + 0x7FFFu + ((u >> 16) & 1u);   // round-to-nearest-even
    return (u16)(r >> 16);
}

__device__ __forceinline__ v8f wmma_bf16(const Frag16& a, const Frag16& b, v8f c) {
    return __builtin_amdgcn_wmma_f32_16x16x32_bf16(
        false, a.bf, false, b.bf, (short)0, c, false, false);
}

// -------- async global -> LDS staging (CDNA5, ASYNCcnt) with sync fallback ---
__device__ __forceinline__ void stage8(u16* lds_dst, const u16* g) {
#if USE_ASYNC
    asm volatile("global_load_async_to_lds_b64 %0, %1, off"
                 :: "v"((u32)(uintptr_t)lds_dst), "v"((u64)(uintptr_t)g)
                 : "memory");
#else
    *(u64*)lds_dst = *(const u64*)g;
#endif
}
__device__ __forceinline__ void stage16(u16* lds_dst, const u16* g) {
#if USE_ASYNC
    asm volatile("global_load_async_to_lds_b128 %0, %1, off"
                 :: "v"((u32)(uintptr_t)lds_dst), "v"((u64)(uintptr_t)g)
                 : "memory");
#else
    *(v4u*)lds_dst = *(const v4u*)g;
#endif
}
__device__ __forceinline__ void stage_wait() {
#if USE_ASYNC
    asm volatile("s_wait_asynccnt 0x0" ::: "memory");
#endif
}

// ---------------------------------------------------------------------------
#define IMG_H 44
#define IMG_W 80
#define NPIX  3520
#define PAD_H 46
#define PAD_W 82
#define NPPIX (PAD_H * PAD_W)     // 3772 padded pixels
#define NVOX  200000
#define VPRE  64
#define DBINS 50
#define C1IN  3200                // 64*50 input channels of conv1
#define K1    28800               // 9 * 3200, k = tap*3200 + (d*64 + c)
#define K2    2304                // 9 * 256,  k = tap*256 + m1
#define M1    256
#define M2    32

// ---------------------------------------------------------------------------
// Kernel Z: zero pf/x1 padded buffers (border ring must be 0 for im2col)
// ---------------------------------------------------------------------------
__global__ __launch_bounds__(256) void zero_k(u64* __restrict__ p, int n) {
    int i = blockIdx.x * 256 + threadIdx.x;
    if (i < n) p[i] = 0ull;
}

// ---------------------------------------------------------------------------
// Kernel 0: weights fp32 -> bf16 with K-permutation.
//   conv1: dst[m][tap*3200 + d*64 + c] = w_r1[m][c*50+d][tap/3][tap%3]
//   conv2: dst[m][tap*256 + ci]        = w_r2[m][ci][tap/3][tap%3]
// ---------------------------------------------------------------------------
__global__ __launch_bounds__(256) void convert_weights_k(
    const float* __restrict__ w1, const float* __restrict__ w2,
    u16* __restrict__ o1, u16* __restrict__ o2)
{
    int i = blockIdx.x * 256 + threadIdx.x;
    if (i < M1 * K1) {
        int m   = i / K1;
        int r   = i - m * K1;
        int tap = r / C1IN;
        int q   = r - tap * C1IN;
        int d   = q >> 6;
        int c   = q & 63;
        int src = (m * C1IN + c * DBINS + d) * 9 + tap;
        o1[i] = f2bf(w1[src]);
    }
    int j = i - M1 * K1;
    if (j >= 0 && j < M2 * K2) {
        int m   = j / K2;
        int r   = j - m * K2;
        int tap = r >> 8;
        int ci  = r & 255;
        int src = (m * 256 + ci) * 9 + tap;
        o2[j] = f2bf(w2[src]);
    }
}

// ---------------------------------------------------------------------------
// Kernel 1: backproject features into voxel grid + fused 1x1 convs
// One wave per voxel. Output voxf [64][200000] fp32 (L2-resident).
// ---------------------------------------------------------------------------
__global__ __launch_bounds__(256) void voxel_fuse_k(
    const float* __restrict__ feats,   // [6][64][44][80]
    const float* __restrict__ maskp,   // [6][1][44][80]
    const float* __restrict__ intr,    // [6][4][4]
    const float* __restrict__ extinv,  // [6][4][4]
    const float* __restrict__ w_no,    // [64][65]
    const float* __restrict__ b_no,    // [64]
    const float* __restrict__ w_o,     // [64][130]
    const float* __restrict__ b_o,     // [64]
    float* __restrict__ voxf)          // [64][200000]
{
    __shared__ float f1s[8][66];
    __shared__ float f2s[8][66];

    const int tid  = threadIdx.x;
    const int lane = tid & 31;
    const int wave = tid >> 5;
    const int v    = blockIdx.x * 8 + wave;

    const int ix = v % 100;
    const int iy = (v / 100) % 100;
    const int iz = v / 10000;
    const float wx = -50.0f + (float)ix;
    const float wy = -50.0f + (float)iy;
    const float wz = -15.0f + 1.5f * (float)iz;

    const int c0 = lane * 2;
    float a1[2] = {0.f, 0.f}, a2[2] = {0.f, 0.f};
    float z1 = 0.f, z2 = 0.f;
    int count = 0;

    #pragma unroll
    for (int cam = 0; cam < 6; ++cam) {
        const float* E = extinv + cam * 16;
        const float lx = E[0]*wx + E[1]*wy + E[2]*wz  + E[3];
        const float ly = E[4]*wx + E[5]*wy + E[6]*wz  + E[7];
        const float lz = E[8]*wx + E[9]*wy + E[10]*wz + E[11];
        const float* Km = intr + cam * 16;
        const float cx = Km[0]*lx + Km[1]*ly + Km[2]*lz;
        const float cy = Km[4]*lx + Km[5]*ly + Km[6]*lz;
        const float cz = Km[8]*lx + Km[9]*ly + Km[10]*lz;
        const float izv = 1.0f / (cz + 1e-8f);
        const float px = cx * izv;
        const float py = cy * izv;

        const bool oob = (px >= 0.f) && (px <= 79.f) && (py >= 0.f) && (py <= 43.f);
        const int nx = (int)floorf(px + 0.5f);
        const int ny = (int)floorf(py + 0.5f);
        bool occ = false;
        if (nx >= 0 && nx <= 79 && ny >= 0 && ny <= 43)
            occ = maskp[(cam * IMG_H + ny) * IMG_W + nx] > 0.5f;
        const bool valid = occ && (lz > 0.f) && oob;

        if (valid) {
            ++count;
            const float x0f = floorf(px), y0f = floorf(py);
            const int x0 = (int)x0f, y0 = (int)y0f;
            const float fx = px - x0f, fy = py - y0f;
            const float w00 = (1.f-fx)*(1.f-fy), w01 = fx*(1.f-fy);
            const float w10 = (1.f-fx)*fy,       w11 = fx*fy;
            const bool vx0 = (x0 >= 0 && x0 <= 79), vx1 = (x0+1 <= 79);
            const bool vy0 = (y0 >= 0 && y0 <= 43), vy1 = (y0+1 <= 43);
            const float* fb = feats + cam * VPRE * NPIX;
            float s[2];
            #pragma unroll
            for (int t = 0; t < 2; ++t) {
                const float* fc = fb + (c0 + t) * NPIX;
                const float p00 = (vx0 && vy0) ? fc[y0*IMG_W + x0]       : 0.f;
                const float p01 = (vx1 && vy0) ? fc[y0*IMG_W + x0+1]     : 0.f;
                const float p10 = (vx0 && vy1) ? fc[(y0+1)*IMG_W + x0]   : 0.f;
                const float p11 = (vx1 && vy1) ? fc[(y0+1)*IMG_W + x0+1] : 0.f;
                s[t] = w00*p00 + w01*p01 + w10*p10 + w11*p11;
            }
            const float zn = lz * 0.01f;
            if (cam == 0 || cam == 3 || cam == 4) {
                a1[0] += s[0]; a1[1] += s[1]; z1 += zn;
            } else {
                a2[0] += s[0]; a2[1] += s[1]; z2 += zn;
            }
        }
    }

    f1s[wave][c0] = a1[0]; f1s[wave][c0+1] = a1[1];
    f2s[wave][c0] = a2[0]; f2s[wave][c0+1] = a2[1];
    if (lane == 0) { f1s[wave][64] = z1; f2s[wave][64] = z2; }
    __syncthreads();

    const float no_m = (count == 1) ? 1.f : 0.f;
    const float ov_m = (count == 2) ? 1.f : 0.f;

    if (no_m + ov_m > 0.f) {
        #pragma unroll
        for (int t = 0; t < 2; ++t) {
            const int o = lane + t * 32;
            float sn = b_no[o];
            float so = b_o[o];
            const float* wn = w_no + o * 65;
            const float* wo = w_o  + o * 130;
            #pragma unroll 5
            for (int c = 0; c < 65; ++c) {
                const float xa = f1s[wave][c];
                const float xb = f2s[wave][c];
                sn = fmaf(wn[c], xa + xb, sn);
                so = fmaf(wo[c], xa, so);
                so = fmaf(wo[65 + c], xb, so);
            }
            const float out = fmaxf(sn, 0.f) * no_m + fmaxf(so, 0.f) * ov_m;
            voxf[o * NVOX + v] = out;
        }
    } else {
        voxf[(lane)      * NVOX + v] = 0.f;
        voxf[(lane + 32) * NVOX + v] = 0.f;
    }
}

// ---------------------------------------------------------------------------
// Kernel 2: trilinear-project voxel volume for one cam into channel-last
// padded pf: pfcl[(py+1)*82 + px+1][d*64 + c] bf16. Wave per (d,pixel) point,
// lanes over channels -> coalesced packed 4B stores. vol gathers hit L2.
// ---------------------------------------------------------------------------
__global__ __launch_bounds__(256) void project_sample_k(
    const float* __restrict__ vox,   // [64][200000]
    const float* __restrict__ invK,  // [6][4][4]
    const float* __restrict__ ext,   // [6][4][4]
    u16* __restrict__ pfcl,          // [3772][3200]
    int cam)
{
    const int tid  = threadIdx.x;
    const int lane = tid & 31;
    const int wave = tid >> 5;
    const int idx  = blockIdx.x * 8 + wave;      // 22000*8 = 176000 exact
    const int d    = idx / NPIX;
    const int p    = idx - d * NPIX;
    const int py   = p / IMG_W;
    const int px   = p - py * IMG_W;

    const float* IK = invK + cam * 16;
    const float rx = IK[0]*px + IK[1]*py + IK[2];
    const float ry = IK[4]*px + IK[5]*py + IK[6];
    const float rz = IK[8]*px + IK[9]*py + IK[10];
    const float t  = 2.0f + (48.0f / 49.0f) * (float)d;
    const float cx = rx * t, cy = ry * t, cz = rz * t;

    const float* E = ext + cam * 16;
    const float wxp = E[0]*cx + E[1]*cy + E[2]*cz  + E[3];
    const float wyp = E[4]*cx + E[5]*cy + E[6]*cz  + E[7];
    const float wzp = E[8]*cx + E[9]*cy + E[10]*cz + E[11];

    const float xs = wxp + 50.f;
    const float ys = wyp + 50.f;
    const float zs = (wzp + 15.f) * (1.0f / 1.5f);

    const float x0f = floorf(xs), y0f = floorf(ys), z0f = floorf(zs);
    const int x0 = (int)x0f, y0 = (int)y0f, z0 = (int)z0f;
    const float fx = xs - x0f, fy = ys - y0f, fz = zs - z0f;

    int   off[8];
    float wt[8];
    #pragma unroll
    for (int c8 = 0; c8 < 8; ++c8) {
        const int dx = c8 & 1, dy = (c8 >> 1) & 1, dz = c8 >> 2;
        const int xi = x0 + dx, yi = y0 + dy, zi = z0 + dz;
        const bool ok = (xi >= 0 && xi <= 99 && yi >= 0 && yi <= 99 && zi >= 0 && zi <= 19);
        const float w = (dx ? fx : 1.f - fx) * (dy ? fy : 1.f - fy) * (dz ? fz : 1.f - fz);
        wt[c8]  = ok ? w : 0.f;
        off[c8] = ok ? ((zi * 100 + yi) * 100 + xi) : 0;
    }

    const int c0 = lane * 2;
    const float* b0 = vox + c0 * NVOX;
    const float* b1 = vox + (c0 + 1) * NVOX;
    float s0 = 0.f, s1 = 0.f;
    #pragma unroll
    for (int i = 0; i < 8; ++i) {
        s0 = fmaf(wt[i], b0[off[i]], s0);
        s1 = fmaf(wt[i], b1[off[i]], s1);
    }

    const int pixp = (py + 1) * PAD_W + (px + 1);
    const u32 pk = (u32)f2bf(s0) | ((u32)f2bf(s1) << 16);
    *(u32*)(pfcl + (size_t)pixp * C1IN + d * 64 + c0) = pk;
}

// ---------------------------------------------------------------------------
// Kernel 3: conv r1 as implicit GEMM, bf16 WMMA, async double-buffered LDS.
// C[256 x 3520] = A'[256 x 28800] * B[28800 x 3520]; K = tap*3200 + ci so each
// 32-wide K-step has uniform tap and B^T rows are contiguous in channel-last
// padded pf -> staged with global_load_async_to_lds_b64. Block: 8 waves,
// tile M=256 x N=32; each wave 2x2 C tiles -> 4 v_wmma per K-step, 900 steps.
// ---------------------------------------------------------------------------
__global__ __launch_bounds__(256) void conv1_wmma_k(
    const u16* __restrict__ A,      // [256][28800] bf16 (permuted)
    const u16* __restrict__ pfcl,   // [3772][3200] bf16 padded channel-last
    const float* __restrict__ bias, // [256]
    u16* __restrict__ x1cl)         // [3772][256] bf16 padded channel-last out
{
    __shared__ __align__(16) u16 lb[2][32][40];   // B^T tiles [n][k], padded rows

    const int tid  = threadIdx.x;
    const int lane = tid & 31;
    const int wave = tid >> 5;
    const int n0   = blockIdx.x * 32;

    // staging role: thread copies 8B = 4 k-halves of row n
    const int snn = tid & 31;
    const int skg = tid >> 5;                       // 0..7
    const int n_g = n0 + snn;
    const int spy = n_g / IMG_W;
    const int spx = n_g - spy * IMG_W;

    // per-tap global row base for this thread's chunk (contiguous in ci)
    const u16* rb[10];
    #pragma unroll
    for (int tap = 0; tap < 9; ++tap) {
        const int ty = tap / 3, tx = tap - 3 * (tap / 3);
        rb[tap] = pfcl + (size_t)((spy + ty) * PAD_W + (spx + tx)) * C1IN + skg * 4;
    }
    rb[9] = rb[0];  // never used (guarded), keeps constant indexing in registers

    u16* ld0 = &lb[0][snn][skg * 4];
    u16* ld1 = &lb[1][snn][skg * 4];

    // compute role
    const int mrow = wave * 32 + (lane & 15);
    const int ko   = (lane < 16) ? 0 : 8;
    const int bkb  = (lane < 16) ? 0 : 8;
    const u16* ap_base0 = A + mrow * K1 + ko;
    const u16* ap_base1 = ap_base0 + 16 * K1;
    const u16* bpA0 = &lb[0][(lane & 15)][bkb];
    const u16* bpA1 = &lb[0][16 + (lane & 15)][bkb];
    const u16* bpB0 = &lb[1][(lane & 15)][bkb];
    const u16* bpB1 = &lb[1][16 + (lane & 15)][bkb];

    v8f acc[2][2];
    {
        v8f z = {0.f,0.f,0.f,0.f,0.f,0.f,0.f,0.f};
        acc[0][0] = z; acc[0][1] = z; acc[1][0] = z; acc[1][1] = z;
    }

    // prologue: stage K-step 0 into buffer 0
    stage8(ld0, rb[0]);
    stage_wait();
    __syncthreads();

    int cur = 0;
    #pragma unroll
    for (int tap = 0; tap < 9; ++tap) {
        #pragma unroll 1
        for (int cs = 0; cs < 100; ++cs) {
            const int ks = tap * 100 + cs;
            // issue next K-step into the other buffer (overlaps with WMMAs)
            if (ks != 899) {
                const u16* nb  = (cs == 99) ? rb[tap + 1] : rb[tap];
                const int  ncs = (cs == 99) ? 0 : cs + 1;
                stage8(cur ? ld0 : ld1, nb + ncs * 32);
            }

            Frag16 a0, a1, b0, b1;
            const u16* ap0 = ap_base0 + ks * 32;
            const u16* ap1 = ap_base1 + ks * 32;
            a0.q[0] = *(const v4u*)(ap0);
            a0.q[1] = *(const v4u*)(ap0 + 16);
            a1.q[0] = *(const v4u*)(ap1);
            a1.q[1] = *(const v4u*)(ap1 + 16);
            __builtin_prefetch(ap0 + 32, 0, 3);

            const u16* bp0 = cur ? bpB0 : bpA0;
            const u16* bp1 = cur ? bpB1 : bpA1;
            b0.q[0] = *(const v4u*)(bp0);
            b0.q[1] = *(const v4u*)(bp0 + 16);
            b1.q[0] = *(const v4u*)(bp1);
            b1.q[1] = *(const v4u*)(bp1 + 16);

            acc[0][0] = wmma_bf16(a0, b0, acc[0][0]);
            acc[0][1] = wmma_bf16(a0, b1, acc[0][1]);
            acc[1][0] = wmma_bf16(a1, b0, acc[1][0]);
            acc[1][1] = wmma_bf16(a1, b1, acc[1][1]);

            stage_wait();        // my async for next step landed
            __syncthreads();     // everyone's landed; cur buffer free next iter
            cur ^= 1;
        }
    }

    // epilogue: bias + ReLU -> packed bf16 x8 store into channel-last x1
    const int mhalf = (lane >> 4) << 3;
    const int ncol  = lane & 15;
    #pragma unroll
    for (int nt = 0; nt < 2; ++nt) {
        const int nn  = n0 + nt * 16 + ncol;
        const int npy = nn / IMG_W;
        const int npx = nn - npy * IMG_W;
        const int pixp = (npy + 1) * PAD_W + (npx + 1);
        #pragma unroll
        for (int mt = 0; mt < 2; ++mt) {
            const int mbase = wave * 32 + mt * 16 + mhalf;
            v4u pk;
            #pragma unroll
            for (int h = 0; h < 4; ++h) {
                float v0 = fmaxf(acc[mt][nt][2*h]   + bias[mbase + 2*h],   0.f);
                float v1 = fmaxf(acc[mt][nt][2*h+1] + bias[mbase + 2*h+1], 0.f);
                pk[h] = (u32)f2bf(v0) | ((u32)f2bf(v1) << 16);
            }
            *(v4u*)(x1cl + (size_t)pixp * 256 + mbase) = pk;
        }
    }
}

// ---------------------------------------------------------------------------
// Kernel 4: conv r2 (256->32) implicit GEMM, bf16 WMMA, async staging,
// fp32 output + ReLU. Block tile M=32 x N=64; one 16x16 C tile per wave.
// K = tap*256 + ci over channel-last padded x1.
// ---------------------------------------------------------------------------
__global__ __launch_bounds__(256) void conv2_wmma_k(
    const u16* __restrict__ A,      // [32][2304] bf16 (permuted)
    const u16* __restrict__ x1cl,   // [3772][256] bf16 padded channel-last
    const float* __restrict__ bias, // [32]
    float* __restrict__ out)        // [32][3520] fp32 (per-cam slice of d_out)
{
    __shared__ __align__(16) u16 lb[2][64][40];

    const int tid  = threadIdx.x;
    const int lane = tid & 31;
    const int wave = tid >> 5;
    const int n0   = blockIdx.x * 64;

    const int snn = tid & 63;
    const int skg = tid >> 6;                  // 0..3, 16B each
    const int n_g = n0 + snn;
    const int spy = n_g / IMG_W;
    const int spx = n_g - spy * IMG_W;

    const u16* rb[10];
    #pragma unroll
    for (int tap = 0; tap < 9; ++tap) {
        const int ty = tap / 3, tx = tap - 3 * (tap / 3);
        rb[tap] = x1cl + (size_t)((spy + ty) * PAD_W + (spx + tx)) * 256 + skg * 8;
    }
    rb[9] = rb[0];

    u16* ld0 = &lb[0][snn][skg * 8];
    u16* ld1 = &lb[1][snn][skg * 8];

    const int mtile = wave >> 2;               // 0..1
    const int ntile = wave & 3;                // 0..3
    const int mrow  = mtile * 16 + (lane & 15);
    const int ko    = (lane < 16) ? 0 : 8;
    const int bkb   = (lane < 16) ? 0 : 8;
    const u16* ap_base = A + mrow * K2 + ko;
    const u16* bpA = &lb[0][ntile * 16 + (lane & 15)][bkb];
    const u16* bpB = &lb[1][ntile * 16 + (lane & 15)][bkb];

    v8f acc = {0.f,0.f,0.f,0.f,0.f,0.f,0.f,0.f};

    stage16(ld0, rb[0]);
    stage_wait();
    __syncthreads();

    int cur = 0;
    #pragma unroll
    for (int tap = 0; tap < 9; ++tap) {
        #pragma unroll 1
        for (int cs = 0; cs < 8; ++cs) {
            const int ks = tap * 8 + cs;
            if (ks != 71) {
                const u16* nb  = (cs == 7) ? rb[tap + 1] : rb[tap];
                const int  ncs = (cs == 7) ? 0 : cs + 1;
                stage16(cur ? ld0 : ld1, nb + ncs * 32);
            }

            Frag16 a, b;
            const u16* ap = ap_base + ks * 32;
            a.q[0] = *(const v4u*)(ap);
            a.q[1] = *(const v4u*)(ap + 16);
            const u16* bp = cur ? bpB : bpA;
            b.q[0] = *(const v4u*)(bp);
            b.q[1] = *(const v4u*)(bp + 16);

            acc = wmma_bf16(a, b, acc);

            stage_wait();
            __syncthreads();
            cur ^= 1;
        }
    }

    const int mhalf = (lane >> 4) << 3;
    const int nn    = n0 + ntile * 16 + (lane & 15);
    #pragma unroll
    for (int r = 0; r < 8; ++r) {
        const int m = mtile * 16 + mhalf + r;
        out[m * NPIX + nn] = fmaxf(acc[r] + bias[m], 0.f);
    }
}

// ---------------------------------------------------------------------------
// Workspace layout (bytes):
//   voxf  [64][200000] f32        : off 0           size 51,200,000
//   pfcl  [3772][3200] bf16       : off 51,200,000  size 24,140,800 (per-cam reuse)
//   x1cl  [3772][256]  bf16       : off 75,340,800  size  1,931,264
//   wr1b  [256][28800] bf16       : off 77,272,064  size 14,745,600
//   wr2b  [32][2304]   bf16       : off 92,017,664  size    147,456
// ---------------------------------------------------------------------------
extern "C" void kernel_launch(void* const* d_in, const int* in_sizes, int n_in,
                              void* d_out, int out_size, void* d_ws, size_t ws_size,
                              hipStream_t stream)
{
    const float* feats  = (const float*)d_in[0];
    const float* maskp  = (const float*)d_in[1];
    const float* intr   = (const float*)d_in[2];
    const float* ext    = (const float*)d_in[3];
    const float* extinv = (const float*)d_in[4];
    const float* invK   = (const float*)d_in[5];
    const float* w_no   = (const float*)d_in[6];
    const float* b_no   = (const float*)d_in[7];
    const float* w_o    = (const float*)d_in[8];
    const float* b_o    = (const float*)d_in[9];
    const float* w_r1   = (const float*)d_in[10];
    const float* b_r1   = (const float*)d_in[11];
    const float* w_r2   = (const float*)d_in[12];
    const float* b_r2   = (const float*)d_in[13];

    char* ws = (char*)d_ws;
    float* voxf = (float*)(ws);
    u16*   pfcl = (u16*)(ws + 51200000);
    u16*   x1cl = (u16*)(ws + 75340800);
    u16*   wr1b = (u16*)(ws + 77272064);
    u16*   wr2b = (u16*)(ws + 92017664);

    // zero pf + x1 padded buffers (adjacent regions; border ring must be 0)
    const int nzero = (24140800 + 1931264) / 8;
    zero_k<<<(nzero + 255) / 256, 256, 0, stream>>>((u64*)pfcl, nzero);

    convert_weights_k<<<(M1*K1 + M2*K2 + 255) / 256, 256, 0, stream>>>(w_r1, w_r2, wr1b, wr2b);

    voxel_fuse_k<<<NVOX / 8, 256, 0, stream>>>(feats, maskp, intr, extinv,
                                               w_no, b_no, w_o, b_o, voxf);

    float* outp = (float*)d_out;
    for (int cam = 0; cam < 6; ++cam) {
        project_sample_k<<<(DBINS * NPIX) / 8, 256, 0, stream>>>(voxf, invK, ext, pfcl, cam);
        conv1_wmma_k<<<NPIX / 32, 256, 0, stream>>>(wr1b, pfcl, b_r1, x1cl);
        conv2_wmma_k<<<NPIX / 64, 256, 0, stream>>>(wr2b, x1cl, b_r2, outp + cam * M2 * NPIX);
    }
}
